// AngleAwareFeatureAlignment_1906965479496
// MI455X (gfx1250) — compile-verified, matrix operand order checked by reading, not run on hardware
//
#include <hip/hip_runtime.h>
#include <hip/hip_bf16.h>

// ---------------------------------------------------------------------------
// AngleAwareFeatureAlignment for MI455X (gfx1250, wave32, WMMA).
// GEMM-shaped work -> v_wmma_f32_16x16x32_f16 with f32 accumulation.
// Weights/activations repacked to f16 so every fragment K-run is one b128 load.
// Weight tiles staged via global_load_async_to_lds_b128 (double buffered,
// shared by 8 waves/WG), synchronized with s_wait_asynccnt + barrier.
// ---------------------------------------------------------------------------

#define HW4   4096
#define NPIX  8192
#define PI_F  3.14159265358979323846f

typedef __attribute__((ext_vector_type(16))) _Float16 v16h;
typedef __attribute__((ext_vector_type(8)))  _Float16 v8h;
typedef __attribute__((ext_vector_type(4)))  _Float16 v4h;
typedef __attribute__((ext_vector_type(8)))  float    v8f;

// K-run base of a 16-bit fragment (16x32, wave32): lanes 0-15 hold
// K {0..7,16..23}; lanes 16-31 hold K {8..15,24..31}.  hi8 = 8*(lane>=16).
__device__ __forceinline__ v16h mk16(v8h lo, v8h hi) {
    v16h r;
    #pragma unroll
    for (int i = 0; i < 8; ++i) { r[i] = lo[i]; r[i + 8] = hi[i]; }
    return r;
}

#define WMMA_F16(acc, a, b) \
    __builtin_amdgcn_wmma_f32_16x16x32_f16(false, (a), false, (b), (short)0, (acc), false, false)

// CDNA5 async load: 16B global -> LDS, tracked by ASYNCcnt.
__device__ __forceinline__ void async_ld_b128(unsigned lds_off, const void* gaddr) {
    asm volatile("global_load_async_to_lds_b128 %0, %1, off"
                 :: "v"(lds_off), "v"(gaddr) : "memory");
}
__device__ __forceinline__ void wait_async0() {
    asm volatile("s_wait_asynccnt 0" ::: "memory");
}
__device__ __forceinline__ unsigned lds_addr_of(const void* p) {
    // flat LDS address keeps the LDS byte offset in the low 32 bits
    return (unsigned)(unsigned long long)p;
}

// ---------------------------------------------------------------------------
// Repack helpers (run once per launch).
// ---------------------------------------------------------------------------
// NCHW f32 -> NHWC f16 (pixels x C row-major), 4 channels per thread.
__global__ __launch_bounds__(256)
void nchw_to_nhwc_h(const float* __restrict__ in, _Float16* __restrict__ out)
{
    const size_t j4 = ((size_t)blockIdx.x * 256 + threadIdx.x) * 4;  // over B*HW*256
    const int b = (int)(j4 >> 20);
    const int p = (int)((j4 >> 8) & 4095);
    const int c = (int)(j4 & 255);
    v4h o;
    #pragma unroll
    for (int i = 0; i < 4; ++i)
        o[i] = (_Float16)in[((size_t)(b * 256 + c + i)) * HW4 + p];
    *(v4h*)(out + j4) = o;
}

// (O,Cin,3,3) f32 -> f16 [tap][O][Cin] (contiguous in Cin), 4 c per thread.
__global__ __launch_bounds__(256)
void pack_w3_h(const float* __restrict__ in, _Float16* __restrict__ out, int O, int Cin)
{
    const size_t i4 = ((size_t)blockIdx.x * 256 + threadIdx.x) * 4;
    if (i4 >= (size_t)O * Cin * 9) return;
    const int tp = (int)(i4 / ((size_t)O * Cin));
    const int r  = (int)(i4 % ((size_t)O * Cin));
    const int o  = r / Cin, c = r % Cin;
    v4h v;
    #pragma unroll
    for (int i = 0; i < 4; ++i)
        v[i] = (_Float16)in[((size_t)o * Cin + c + i) * 9 + tp];
    *(v4h*)(out + i4) = v;
}

// plain f32 -> f16 (n multiple of 4)
__global__ __launch_bounds__(256)
void cvt_h(const float* __restrict__ in, _Float16* __restrict__ out, int n)
{
    const int i4 = (blockIdx.x * 256 + threadIdx.x) * 4;
    if (i4 >= n) return;
    const float4 f = *(const float4*)(in + i4);
    v4h v; v[0] = (_Float16)f.x; v[1] = (_Float16)f.y; v[2] = (_Float16)f.z; v[3] = (_Float16)f.w;
    *(v4h*)(out + i4) = v;
}

// ---------------------------------------------------------------------------
// 3x3 conv (pad=1) implicit GEMM, input NHWC f16, weights packed [tap][O][Cin] f16.
// 8 waves / WG: wave w -> M-tile, shared 64-wide N tile staged async into LDS.
// Output f32 row-major (pixels x Cout)  [== NHWC].
// ---------------------------------------------------------------------------
__global__ __launch_bounds__(256)
void conv3x3_wmma_k(const _Float16* __restrict__ in, const _Float16* __restrict__ Wp,
                    const float* __restrict__ bias, float* __restrict__ out,
                    int Cout, int relu)
{
    __shared__ __align__(16) _Float16 lB[2][64 * 32];
    const int tid = threadIdx.x;
    const int lane = tid & 31, w = tid >> 5;
    const int mloc = lane & 15;
    const int hi8 = (lane >> 4) << 3;
    const int mt = blockIdx.x * 8 + w, nb = blockIdx.y;
    const int m = mt * 16 + mloc;
    const int bi = m >> 12, p = m & 4095;
    const int y = p >> 6, x = p & 63;

    const int sn = tid >> 2, srun = tid & 3;      // staging slice: n, k-run
    const unsigned lbase = lds_addr_of(&lB[0][0]);
    auto wsrc = [&](int q) -> const void* {       // q = tap*8 + cc  (Cin=256)
        const int tp = q >> 3, cc = q & 7;
        return Wp + ((size_t)(tp * Cout + nb * 64 + sn)) * 256 + cc * 32 + srun * 8;
    };
    async_ld_b128(lbase + tid * 16u, wsrc(0));
    __builtin_prefetch(in + (size_t)(bi * HW4 + y * 64 + x) * 256, 0, 0);

    v8f acc[4] = {};
    for (int q = 0; q < 72; ++q) {
        wait_async0();
        __syncthreads();
        if (q + 1 < 72)
            async_ld_b128(lbase + ((q + 1) & 1) * 4096u + tid * 16u, wsrc(q + 1));

        const int tp = q >> 3, cc = q & 7, c0 = cc << 5;
        const int dy = tp / 3 - 1, dx = tp % 3 - 1;
        const int yy = y + dy, xx = x + dx;
        const bool ok = (yy >= 0 && yy < 64 && xx >= 0 && xx < 64);
        v16h a;
        if (ok) {
            const _Float16* arow = in + ((size_t)(bi * HW4 + yy * 64 + xx)) * 256;
            a = mk16(*(const v8h*)(arow + c0 + hi8), *(const v8h*)(arow + c0 + 16 + hi8));
        } else {
            a = (v16h)(_Float16)0.f;
        }
        const _Float16* lb = &lB[q & 1][0];
        #pragma unroll
        for (int nt = 0; nt < 4; ++nt) {
            const int nl = nt * 16 + mloc;
            const v16h b = mk16(*(const v8h*)(lb + nl * 32 + hi8),
                                *(const v8h*)(lb + nl * 32 + 16 + hi8));
            acc[nt] = WMMA_F16(acc[nt], a, b);
        }
    }
    #pragma unroll
    for (int nt = 0; nt < 4; ++nt) {
        const int n = nb * 64 + nt * 16 + mloc;
        const float bv = bias[n];
        #pragma unroll
        for (int r = 0; r < 8; ++r) {
            const int mm = mt * 16 + r + hi8;
            float v = acc[nt][r] + bv;
            if (relu) v = fmaxf(v, 0.f);
            out[(size_t)mm * Cout + n] = v;
        }
    }
}

// ---------------------------------------------------------------------------
// angles[m] = dot(h1[m,0:64], w2) + b2   (h1 f32 row-major)
// ---------------------------------------------------------------------------
__global__ __launch_bounds__(256)
void angle_head_k(const float* __restrict__ h1, const float* __restrict__ w2,
                  const float* __restrict__ b2, float* __restrict__ angles)
{
    const int m = blockIdx.x * 256 + threadIdx.x;
    const float4* r = (const float4*)(h1 + (size_t)m * 64);
    const float4* wv = (const float4*)w2;
    float s = b2[0];
    #pragma unroll
    for (int i = 0; i < 16; ++i) {
        const float4 a = r[i], b = wv[i];
        s += a.x * b.x + a.y * b.y + a.z * b.z + a.w * b.w;
    }
    angles[m] = s;
}

// ---------------------------------------------------------------------------
// Periodic angle encoding + MLP(64->64, LN, relu, 64->64); wave / 16 pixels.
// Weights pre-converted f16; output f16 (pixels x 64).
// ---------------------------------------------------------------------------
__global__ __launch_bounds__(32)
void angle_encode_k(const float* __restrict__ angles, const float* __restrict__ fm,
                    const _Float16* __restrict__ w1, const float* __restrict__ b1,
                    const float* __restrict__ lng, const float* __restrict__ lnb,
                    const _Float16* __restrict__ w2, const float* __restrict__ b2,
                    _Float16* __restrict__ aout)
{
    __shared__ float red0[16][16], red1[16][16], mv[16][2];
    __shared__ __align__(16) _Float16 hbuf[16][80];
    const int lane = threadIdx.x & 31;
    const int mloc = lane & 15;
    const int hi8 = (lane >> 4) << 3;
    const int m = blockIdx.x * 16 + mloc;
    const float an = angles[m] * (2.f / PI_F);

    v8f acc[4] = {};
    for (int cc = 0; cc < 2; ++cc) {
        const int c0 = cc << 5;
        v16h a;
        #pragma unroll
        for (int e = 0; e < 16; ++e) {
            const int k = c0 + ((e >> 3) << 4) + hi8 + (e & 7);
            const int j = k & 31;
            const float ae = an * fm[j] * (PI_F * (float)j / 32.f);
            a[e] = (_Float16)((k < 32) ? __sinf(ae) : __cosf(ae));
        }
        #pragma unroll
        for (int nt = 0; nt < 4; ++nt) {
            const _Float16* wr = w1 + (size_t)(nt * 16 + mloc) * 64 + c0;
            acc[nt] = WMMA_F16(acc[nt], a,
                               mk16(*(const v8h*)(wr + hi8), *(const v8h*)(wr + 16 + hi8)));
        }
    }
    #pragma unroll
    for (int nt = 0; nt < 4; ++nt) {
        const float bv = b1[nt * 16 + mloc];
        #pragma unroll
        for (int r = 0; r < 8; ++r) acc[nt][r] += bv;
    }
    #pragma unroll
    for (int r = 0; r < 8; ++r) {
        float s = 0.f, q = 0.f;
        #pragma unroll
        for (int nt = 0; nt < 4; ++nt) { const float v = acc[nt][r]; s += v; q += v * v; }
        red0[r + hi8][mloc] = s; red1[r + hi8][mloc] = q;
    }
    __syncthreads();
    if (lane < 16) {
        float s = 0.f, q = 0.f;
        #pragma unroll
        for (int i = 0; i < 16; ++i) { s += red0[lane][i]; q += red1[lane][i]; }
        const float mu = s / 64.f;
        mv[lane][0] = mu;
        mv[lane][1] = rsqrtf(q / 64.f - mu * mu + 1e-5f);
    }
    __syncthreads();
    #pragma unroll
    for (int nt = 0; nt < 4; ++nt) {
        const int n = nt * 16 + mloc;
        const float g = lng[n], bb = lnb[n];
        #pragma unroll
        for (int r = 0; r < 8; ++r) {
            const int mrow = r + hi8;
            const float v = (acc[nt][r] - mv[mrow][0]) * mv[mrow][1] * g + bb;
            hbuf[mrow][n] = (_Float16)fmaxf(v, 0.f);
        }
    }
    __syncthreads();
    v8f acc2[4] = {};
    for (int cc = 0; cc < 2; ++cc) {
        const int c0 = cc << 5;
        const v16h a = mk16(*(const v8h*)(&hbuf[mloc][c0 + hi8]),
                            *(const v8h*)(&hbuf[mloc][c0 + 16 + hi8]));
        #pragma unroll
        for (int nt = 0; nt < 4; ++nt) {
            const _Float16* wr = w2 + (size_t)(nt * 16 + mloc) * 64 + c0;
            acc2[nt] = WMMA_F16(acc2[nt], a,
                                mk16(*(const v8h*)(wr + hi8), *(const v8h*)(wr + 16 + hi8)));
        }
    }
    #pragma unroll
    for (int nt = 0; nt < 4; ++nt) {
        const int n = nt * 16 + mloc;
        const float bv = b2[n];
        #pragma unroll
        for (int r = 0; r < 8; ++r)
            aout[(size_t)(blockIdx.x * 16 + r + hi8) * 64 + n] = (_Float16)(acc2[nt][r] + bv);
    }
}

// ---------------------------------------------------------------------------
// Rotation-aware deformable 3x3 conv (256->256) implicit GEMM on NHWC f16.
// Bilinear taps: py = y + kx*sin + ky*cos ; px = x + kx*cos - ky*sin.
// Weights packed [tap][256][256] f16, async-staged into LDS (as conv3x3).
// Output f16 row-major (pixels x 256).
// ---------------------------------------------------------------------------
__global__ __launch_bounds__(256)
void deform_wmma_k(const _Float16* __restrict__ feat, const float* __restrict__ angles,
                   float delta, const _Float16* __restrict__ Wp,
                   const float* __restrict__ bias, _Float16* __restrict__ out)
{
    __shared__ __align__(16) _Float16 lB[2][64 * 32];
    const int tid = threadIdx.x;
    const int lane = tid & 31, w = tid >> 5;
    const int mloc = lane & 15;
    const int hi8 = (lane >> 4) << 3;
    const int mt = blockIdx.x * 8 + w, nb = blockIdx.y;
    const int m = mt * 16 + mloc;
    const int bi = m >> 12, p = m & 4095;
    const int y = p >> 6, x = p & 63;
    const _Float16* fb = feat + (size_t)bi * HW4 * 256;
    const float ang = angles[m] + delta;
    const float sn = __sinf(ang), cs = __cosf(ang);

    const int snn = tid >> 2, srun = tid & 3;
    const unsigned lbase = lds_addr_of(&lB[0][0]);
    auto wsrc = [&](int q) -> const void* {
        const int tp = q >> 3, cc = q & 7;
        return Wp + ((size_t)(tp * 256 + nb * 64 + snn)) * 256 + cc * 32 + srun * 8;
    };
    async_ld_b128(lbase + tid * 16u, wsrc(0));

    float w00 = 0.f, w01 = 0.f, w10 = 0.f, w11 = 0.f;
    int o00 = 0, o01 = 0, o10 = 0, o11 = 0;
    v8f acc[4] = {};
    for (int q = 0; q < 72; ++q) {
        wait_async0();
        __syncthreads();
        if (q + 1 < 72)
            async_ld_b128(lbase + ((q + 1) & 1) * 4096u + tid * 16u, wsrc(q + 1));

        const int tp = q >> 3, cc = q & 7, c0 = cc << 5;
        if (cc == 0) {
            const float ky = (float)(tp / 3 - 1), kx = (float)(tp % 3 - 1);
            const float py = (float)y + kx * sn + ky * cs;
            const float px = (float)x + kx * cs - ky * sn;
            const float fy = floorf(py), fx = floorf(px);
            const int iy0 = (int)fy, ix0 = (int)fx;
            const float wy = py - fy, wx = px - fx;
            w00 = (1.f - wy) * (1.f - wx); w01 = (1.f - wy) * wx;
            w10 = wy * (1.f - wx);         w11 = wy * wx;
            auto corner = [](int yy, int xx, float& ww) -> int {
                if (yy < 0 || yy > 63 || xx < 0 || xx > 63) { ww = 0.f; return 0; }
                return yy * 64 + xx;
            };
            o00 = corner(iy0,     ix0,     w00);
            o01 = corner(iy0,     ix0 + 1, w01);
            o10 = corner(iy0 + 1, ix0,     w10);
            o11 = corner(iy0 + 1, ix0 + 1, w11);
        }
        v16h a;
        #pragma unroll
        for (int half = 0; half < 2; ++half) {
            const int kb = c0 + half * 16 + hi8;
            const v8h s00 = *(const v8h*)(fb + (size_t)o00 * 256 + kb);
            const v8h s01 = *(const v8h*)(fb + (size_t)o01 * 256 + kb);
            const v8h s10 = *(const v8h*)(fb + (size_t)o10 * 256 + kb);
            const v8h s11 = *(const v8h*)(fb + (size_t)o11 * 256 + kb);
            #pragma unroll
            for (int i = 0; i < 8; ++i)
                a[half * 8 + i] = (_Float16)(w00 * (float)s00[i] + w01 * (float)s01[i] +
                                             w10 * (float)s10[i] + w11 * (float)s11[i]);
        }
        const _Float16* lb = &lB[q & 1][0];
        #pragma unroll
        for (int nt = 0; nt < 4; ++nt) {
            const int nl = nt * 16 + mloc;
            acc[nt] = WMMA_F16(acc[nt], a,
                               mk16(*(const v8h*)(lb + nl * 32 + hi8),
                                    *(const v8h*)(lb + nl * 32 + 16 + hi8)));
        }
    }
    #pragma unroll
    for (int nt = 0; nt < 4; ++nt) {
        const int n = nb * 64 + nt * 16 + mloc;
        const float bv = bias[n];
        #pragma unroll
        for (int r = 0; r < 8; ++r)
            out[(size_t)(mt * 16 + r + hi8) * 256 + n] = (_Float16)(acc[nt][r] + bv);
    }
}

// ---------------------------------------------------------------------------
// Generic WMMA GEMM, N=256: out[m,n] = act( sum_k A[m,k]*W[n,k] + bias[n] ).
// A f16 row-major (concat: k<256 from A0 width 256, else A1 width 64).
// Weights f16 (N,K) row-major, async-staged to LDS.
// omode 0: f16 row-major; omode 1: f32 NCHW. act 2: out = gsrc*(1+sigmoid(v)).
// ---------------------------------------------------------------------------
__global__ __launch_bounds__(256)
void gemm_wmma_k(const _Float16* __restrict__ A0, const _Float16* __restrict__ A1,
                 const _Float16* __restrict__ Wp, const float* __restrict__ bias,
                 _Float16* __restrict__ outh, float* __restrict__ outf,
                 const _Float16* __restrict__ gsrc,
                 int K, int concat, int omode, int act)
{
    __shared__ __align__(16) _Float16 lB[2][64 * 32];
    const int tid = threadIdx.x;
    const int lane = tid & 31, w = tid >> 5;
    const int mloc = lane & 15;
    const int hi8 = (lane >> 4) << 3;
    const int mt = blockIdx.x * 8 + w, nb = blockIdx.y;
    const int m = mt * 16 + mloc;
    const int ccs = K >> 5;

    const int sn = tid >> 2, srun = tid & 3;
    const _Float16* wrow = Wp + (size_t)(nb * 64 + sn) * K + srun * 8;
    const unsigned lbase = lds_addr_of(&lB[0][0]);
    async_ld_b128(lbase + tid * 16u, wrow);
    __builtin_prefetch(A0 + (size_t)m * (concat ? 256 : K), 0, 0);

    v8f acc[4] = {};
    for (int cc = 0; cc < ccs; ++cc) {
        wait_async0();
        __syncthreads();
        if (cc + 1 < ccs)
            async_ld_b128(lbase + ((cc + 1) & 1) * 4096u + tid * 16u, wrow + (size_t)(cc + 1) * 32);

        const int c0 = cc << 5;
        const _Float16* arow;
        int kk;
        if (!concat || c0 < 256) { arow = A0 + (size_t)m * (concat ? 256 : K); kk = c0; }
        else                     { arow = A1 + (size_t)m * 64;                 kk = c0 - 256; }
        const v16h a = mk16(*(const v8h*)(arow + kk + hi8),
                            *(const v8h*)(arow + kk + 16 + hi8));
        const _Float16* lb = &lB[cc & 1][0];
        #pragma unroll
        for (int nt = 0; nt < 4; ++nt) {
            const int nl = nt * 16 + mloc;
            acc[nt] = WMMA_F16(acc[nt], a,
                               mk16(*(const v8h*)(lb + nl * 32 + hi8),
                                    *(const v8h*)(lb + nl * 32 + 16 + hi8)));
        }
    }
    #pragma unroll
    for (int nt = 0; nt < 4; ++nt) {
        const int n = nb * 64 + nt * 16 + mloc;
        const float bv = bias[n];
        #pragma unroll
        for (int r = 0; r < 8; ++r) {
            const int mm = mt * 16 + r + hi8;
            float v = acc[nt][r] + bv;
            if (act == 1) v = fmaxf(v, 0.f);
            else if (act == 2) {
                const float fz = (float)gsrc[(size_t)mm * 256 + n];
                v = fz * (1.f + 1.f / (1.f + __expf(-v)));
            }
            if (omode == 0) outh[(size_t)mm * 256 + n] = (_Float16)v;
            else {
                const int bo = mm >> 12, po = mm & 4095;
                outf[((size_t)bo * 256 + n) * HW4 + po] = v;
            }
        }
    }
}

// ---------------------------------------------------------------------------
// Per-pixel MHA over 3 hypotheses (1 query). Wave per pixel; lane owns 8
// contiguous channels (one head = 4 lanes); head-group shuffle reduction.
// ---------------------------------------------------------------------------
__global__ __launch_bounds__(256)
void attn_k(const _Float16* __restrict__ Q,
            const _Float16* __restrict__ K0, const _Float16* __restrict__ K1,
            const _Float16* __restrict__ K2,
            const _Float16* __restrict__ V0, const _Float16* __restrict__ V1,
            const _Float16* __restrict__ V2, _Float16* __restrict__ ctx)
{
    const int lane = threadIdx.x & 31;
    const int m = blockIdx.x * 8 + (threadIdx.x >> 5);
    const size_t base = (size_t)m * 256 + lane * 8;
    const v8h q  = *(const v8h*)(Q + base);
    const v8h k0 = *(const v8h*)(K0 + base), k1 = *(const v8h*)(K1 + base), k2 = *(const v8h*)(K2 + base);
    const v8h v0 = *(const v8h*)(V0 + base), v1 = *(const v8h*)(V1 + base), v2 = *(const v8h*)(V2 + base);

    float s0 = 0.f, s1 = 0.f, s2 = 0.f;
    #pragma unroll
    for (int i = 0; i < 8; ++i) {
        const float qv = (float)q[i];
        s0 += qv * (float)k0[i]; s1 += qv * (float)k1[i]; s2 += qv * (float)k2[i];
    }
    // reduce across the 4 lanes of this head (32 dims)
    s0 += __shfl_xor(s0, 1, 32); s0 += __shfl_xor(s0, 2, 32);
    s1 += __shfl_xor(s1, 1, 32); s1 += __shfl_xor(s1, 2, 32);
    s2 += __shfl_xor(s2, 1, 32); s2 += __shfl_xor(s2, 2, 32);

    const float scale = 0.17677669529663687f;  // 1/sqrt(32)
    const float mx = fmaxf(fmaxf(s0, s1), s2) * scale;
    const float e0 = __expf(s0 * scale - mx);
    const float e1 = __expf(s1 * scale - mx);
    const float e2 = __expf(s2 * scale - mx);
    const float inv = 1.f / (e0 + e1 + e2);
    const float a0 = e0 * inv, a1 = e1 * inv, a2 = e2 * inv;

    v8h o;
    #pragma unroll
    for (int i = 0; i < 8; ++i)
        o[i] = (_Float16)(a0 * (float)v0[i] + a1 * (float)v1[i] + a2 * (float)v2[i]);
    *(v8h*)(ctx + base) = o;
}

// ---------------------------------------------------------------------------
// GroupNorm(32) on f32 NHWC (pixels x 256): stats then apply(+relu) -> f16.
// ---------------------------------------------------------------------------
__global__ __launch_bounds__(256)
void gn_stats_k(const float* __restrict__ x, float* __restrict__ st)
{
    const int bg = blockIdx.x;                 // b*32 + g
    const int b = bg >> 5, g = bg & 31;
    float s = 0.f, q = 0.f;
    for (int pp = threadIdx.x; pp < HW4; pp += 256) {
        const float4* r = (const float4*)(x + ((size_t)(b * HW4 + pp)) * 256 + g * 8);
        const float4 u = r[0], v = r[1];
        s += u.x + u.y + u.z + u.w + v.x + v.y + v.z + v.w;
        q += u.x*u.x + u.y*u.y + u.z*u.z + u.w*u.w + v.x*v.x + v.y*v.y + v.z*v.z + v.w*v.w;
    }
    __shared__ float rs[256], rq[256];
    rs[threadIdx.x] = s; rq[threadIdx.x] = q;
    __syncthreads();
    for (int off = 128; off; off >>= 1) {
        if (threadIdx.x < off) { rs[threadIdx.x] += rs[threadIdx.x + off]; rq[threadIdx.x] += rq[threadIdx.x + off]; }
        __syncthreads();
    }
    if (threadIdx.x == 0) {
        const float mu = rs[0] / 32768.f;
        st[bg * 2] = mu;
        st[bg * 2 + 1] = rsqrtf(rq[0] / 32768.f - mu * mu + 1e-5f);
    }
}

__global__ __launch_bounds__(256)
void gn_apply_k(const float* __restrict__ x, const float* __restrict__ st,
                const float* __restrict__ g, const float* __restrict__ bta,
                _Float16* __restrict__ o)
{
    const size_t i4 = ((size_t)blockIdx.x * 256 + threadIdx.x) * 4;   // over 2M, NHWC
    const int c = (int)(i4 & 255);
    const int b = (int)(i4 >> 20);
    const int grp = c >> 3;
    const float mu = st[(b * 32 + grp) * 2], rstd = st[(b * 32 + grp) * 2 + 1];
    const float4 xv = *(const float4*)(x + i4);
    v4h ov;
    ov[0] = (_Float16)fmaxf((xv.x - mu) * rstd * g[c + 0] + bta[c + 0], 0.f);
    ov[1] = (_Float16)fmaxf((xv.y - mu) * rstd * g[c + 1] + bta[c + 1], 0.f);
    ov[2] = (_Float16)fmaxf((xv.z - mu) * rstd * g[c + 2] + bta[c + 2], 0.f);
    ov[3] = (_Float16)fmaxf((xv.w - mu) * rstd * g[c + 3] + bta[c + 3], 0.f);
    *(v4h*)(o + i4) = ov;
}

// ---------------------------------------------------------------------------
extern "C" void kernel_launch(void* const* d_in, const int* in_sizes, int n_in,
                              void* d_out, int out_size, void* d_ws, size_t ws_size,
                              hipStream_t stream) {
    const float* features  = (const float*)d_in[0];
    const float* ap_w1     = (const float*)d_in[1];
    const float* ap_b1     = (const float*)d_in[2];
    const float* ap_w2     = (const float*)d_in[3];
    const float* ap_b2     = (const float*)d_in[4];
    const float* freq_mult = (const float*)d_in[5];
    const float* ae_w1     = (const float*)d_in[6];
    const float* ae_b1     = (const float*)d_in[7];
    const float* ae_ln_g   = (const float*)d_in[8];
    const float* ae_ln_b   = (const float*)d_in[9];
    const float* ae_w2     = (const float*)d_in[10];
    const float* ae_b2     = (const float*)d_in[11];
    const float* dc_w      = (const float*)d_in[12];
    const float* dc_b      = (const float*)d_in[13];
    const float* attn_in_w = (const float*)d_in[14];
    const float* attn_in_b = (const float*)d_in[15];
    const float* attn_out_w= (const float*)d_in[16];
    const float* attn_out_b= (const float*)d_in[17];
    const float* ja_w1     = (const float*)d_in[18];
    const float* ja_b1     = (const float*)d_in[19];
    const float* ja_w2     = (const float*)d_in[20];
    const float* ja_b2     = (const float*)d_in[21];
    const float* op_w1     = (const float*)d_in[22];
    const float* op_b1     = (const float*)d_in[23];
    const float* gn_g      = (const float*)d_in[24];
    const float* gn_b      = (const float*)d_in[25];
    const float* op_w2     = (const float*)d_in[26];
    const float* op_b2     = (const float*)d_in[27];
    float* outp = (float*)d_out;

    // ---- workspace: small f32 region + f16 region (~82 MB total) ----------
    float* wsf    = (float*)d_ws;
    float* angles = wsf;                         // 8192
    float* h1     = angles + 8192;               // NPIX*64
    float* c3out  = h1 + (size_t)NPIX * 64;      // NPIX*256
    float* gnst   = c3out + (size_t)NPIX * 256;  // 128
    _Float16* wh  = (_Float16*)(gnst + 128);
    size_t o = 0;
    const size_t TEN = (size_t)NPIX * 256;
    _Float16* featT = wh + o; o += TEN;
    _Float16* aembh = wh + o; o += (size_t)NPIX * 64;
    _Float16* cand0 = wh + o; o += TEN;
    _Float16* cand1 = wh + o; o += TEN;
    _Float16* cand2 = wh + o; o += TEN;
    _Float16* Qh  = wh + o; o += TEN;
    _Float16* Kh0 = wh + o; o += TEN;
    _Float16* Kh1 = wh + o; o += TEN;
    _Float16* Kh2 = wh + o; o += TEN;
    _Float16* Vh0 = wh + o; o += TEN;
    _Float16* Vh1 = wh + o; o += TEN;
    _Float16* Vh2 = wh + o; o += TEN;
    _Float16* ctxh = wh + o; o += TEN;
    _Float16* fusedh = wh + o; o += TEN;
    _Float16* g1h = wh + o; o += TEN;
    _Float16* enhh = wh + o; o += TEN;
    _Float16* obufh = wh + o; o += TEN;
    _Float16* pw_ap1 = wh + o; o += (size_t)9 * 64 * 256;
    _Float16* pw_dc  = wh + o; o += (size_t)9 * 256 * 256;
    _Float16* pw_op1 = wh + o; o += (size_t)9 * 256 * 256;
    _Float16* pw_qkv = wh + o; o += (size_t)768 * 256;
    _Float16* pw_ao  = wh + o; o += (size_t)256 * 256;
    _Float16* pw_ja1 = wh + o; o += (size_t)256 * 320;
    _Float16* pw_ja2 = wh + o; o += (size_t)256 * 256;
    _Float16* pw_op2 = wh + o; o += (size_t)256 * 256;
    _Float16* pw_ae1 = wh + o; o += (size_t)64 * 64;
    _Float16* pw_ae2 = wh + o; o += (size_t)64 * 64;

    // ---- one-time repacks -------------------------------------------------
    nchw_to_nhwc_h<<<(2 * 256 * HW4) / 1024, 256, 0, stream>>>(features, featT);
    pack_w3_h<<<(9 * 64 * 256) / 1024, 256, 0, stream>>>(ap_w1, pw_ap1, 64, 256);
    pack_w3_h<<<(9 * 256 * 256) / 1024, 256, 0, stream>>>(dc_w, pw_dc, 256, 256);
    pack_w3_h<<<(9 * 256 * 256) / 1024, 256, 0, stream>>>(op_w1, pw_op1, 256, 256);
    cvt_h<<<(768 * 256) / 1024, 256, 0, stream>>>(attn_in_w, pw_qkv, 768 * 256);
    cvt_h<<<(256 * 256) / 1024, 256, 0, stream>>>(attn_out_w, pw_ao, 256 * 256);
    cvt_h<<<(256 * 320) / 1024, 256, 0, stream>>>(ja_w1, pw_ja1, 256 * 320);
    cvt_h<<<(256 * 256) / 1024, 256, 0, stream>>>(ja_w2, pw_ja2, 256 * 256);
    cvt_h<<<(256 * 256) / 1024, 256, 0, stream>>>(op_w2, pw_op2, 256 * 256);
    cvt_h<<<(64 * 64) / 1024 + 1, 256, 0, stream>>>(ae_w1, pw_ae1, 64 * 64);
    cvt_h<<<(64 * 64) / 1024 + 1, 256, 0, stream>>>(ae_w2, pw_ae2, 64 * 64);

    const _Float16* wq = pw_qkv;             const float* bq = attn_in_b;
    const _Float16* wk = pw_qkv + 256 * 256; const float* bk = attn_in_b + 256;
    const _Float16* wv = pw_qkv + 512 * 256; const float* bv = attn_in_b + 512;

    const dim3 blk(256);
    const dim3 gc1(NPIX / 128, 1), gc4(NPIX / 128, 4);

    // 1) angle prediction
    conv3x3_wmma_k<<<gc1, blk, 0, stream>>>(featT, pw_ap1, ap_b1, h1, 64, 1);
    angle_head_k<<<NPIX / 256, blk, 0, stream>>>(h1, ap_w2, ap_b2, angles);
    // 2) periodic angle encoding
    angle_encode_k<<<NPIX / 16, 32, 0, stream>>>(angles, freq_mult, pw_ae1, ae_b1,
                                                 ae_ln_g, ae_ln_b, pw_ae2, ae_b2, aembh);
    // 3/4) deformable hypotheses (delta = 0 is also "aligned")
    const float d12 = PI_F / 12.f;
    deform_wmma_k<<<gc4, blk, 0, stream>>>(featT, angles, -d12, pw_dc, dc_b, cand0);
    deform_wmma_k<<<gc4, blk, 0, stream>>>(featT, angles,  0.f, pw_dc, dc_b, cand1);
    deform_wmma_k<<<gc4, blk, 0, stream>>>(featT, angles,  d12, pw_dc, dc_b, cand2);
    // 4) QKV projections + per-pixel attention + output projection
    gemm_wmma_k<<<gc4, blk, 0, stream>>>(cand1, nullptr, wq, bq, Qh,  nullptr, nullptr, 256, 0, 0, 0);
    gemm_wmma_k<<<gc4, blk, 0, stream>>>(cand0, nullptr, wk, bk, Kh0, nullptr, nullptr, 256, 0, 0, 0);
    gemm_wmma_k<<<gc4, blk, 0, stream>>>(cand1, nullptr, wk, bk, Kh1, nullptr, nullptr, 256, 0, 0, 0);
    gemm_wmma_k<<<gc4, blk, 0, stream>>>(cand2, nullptr, wk, bk, Kh2, nullptr, nullptr, 256, 0, 0, 0);
    gemm_wmma_k<<<gc4, blk, 0, stream>>>(cand0, nullptr, wv, bv, Vh0, nullptr, nullptr, 256, 0, 0, 0);
    gemm_wmma_k<<<gc4, blk, 0, stream>>>(cand1, nullptr, wv, bv, Vh1, nullptr, nullptr, 256, 0, 0, 0);
    gemm_wmma_k<<<gc4, blk, 0, stream>>>(cand2, nullptr, wv, bv, Vh2, nullptr, nullptr, 256, 0, 0, 0);
    attn_k<<<NPIX / 8, blk, 0, stream>>>(Qh, Kh0, Kh1, Kh2, Vh0, Vh1, Vh2, ctxh);
    gemm_wmma_k<<<gc4, blk, 0, stream>>>(ctxh, nullptr, pw_ao, attn_out_b, fusedh,
                                         nullptr, nullptr, 256, 0, 0, 0);
    // 5) joint angle-spatial gate
    gemm_wmma_k<<<gc4, blk, 0, stream>>>(fusedh, aembh, pw_ja1, ja_b1, g1h,
                                         nullptr, nullptr, 320, 1, 0, 1);
    gemm_wmma_k<<<gc4, blk, 0, stream>>>(g1h, nullptr, pw_ja2, ja_b2, enhh,
                                         nullptr, fusedh, 256, 0, 0, 2);
    // 6) output projection: conv3x3 -> GN -> relu -> conv1x1 (f32 NCHW out)
    conv3x3_wmma_k<<<gc4, blk, 0, stream>>>(enhh, pw_op1, op_b1, c3out, 256, 0);
    gn_stats_k<<<64, blk, 0, stream>>>(c3out, gnst);
    gn_apply_k<<<(2 * 256 * HW4) / 1024, blk, 0, stream>>>(c3out, gnst, gn_g, gn_b, obufh);
    gemm_wmma_k<<<gc4, blk, 0, stream>>>(obufh, nullptr, pw_op2, op_b2, nullptr,
                                         outp, nullptr, 256, 0, 1, 0);
}
